// PatchTransformer_81784767250920
// MI455X (gfx1250) — matrix operand order repaired and checked by reading, not compiled, not run on hardware
//
#include <hip/hip_runtime.h>

// ---------------------------------------------------------------------------
// Problem constants (from the reference)
// ---------------------------------------------------------------------------
#define DIMC   512
#define HEADSC 8
#define HDC    64
#define NPATCH 100
#define SSEQ   101          // N + 1
#define BATCH  2
#define PDC    256
#define MROWS  (NPATCH * SSEQ * BATCH)   // 20200 token rows

// ---------------------------------------------------------------------------
// Types for CDNA5 WMMA (wave32, 16x16x32 bf16 -> f32)
// ---------------------------------------------------------------------------
typedef __attribute__((ext_vector_type(16))) __bf16 bf16x16;
typedef __attribute__((ext_vector_type(8)))  float  f32x8;
typedef __attribute__((__vector_size__(16))) int    i32x4;

union AFrag { bf16x16 v; unsigned int u[8]; };

__device__ __forceinline__ unsigned short f2bf(float f) {
  unsigned int u = __float_as_uint(f);
  u += 0x7FFFu + ((u >> 16) & 1u);          // round-to-nearest-even
  return (unsigned short)(u >> 16);
}
__device__ __forceinline__ float bf2f(unsigned short u) {
  return __uint_as_float(((unsigned int)u) << 16);
}

// ---------------------------------------------------------------------------
// Generic bf16 WMMA GEMM:  Out[M,N] = A[M,K] * W[N,K]^T + bias
//   A : bf16 row-major (ld K); W : bf16 row-major (N rows x K)
//   256 thr = 8 waves (2x4), block tile 64(M) x 128(N),
//   wave tile 32x32 via 2x2 v_wmma_f32_16x16x32_bf16, K step 32.
// ---------------------------------------------------------------------------
template<int RELU, int WF32, int WBF16>
__global__ __launch_bounds__(256)
void wmma_gemm_bf16(const unsigned short* __restrict__ A,
                    const unsigned short* __restrict__ W,
                    const float* __restrict__ bias,
                    float* __restrict__ Of32,
                    unsigned short* __restrict__ Obf,
                    int M, int N, int K)
{
  const int lane = threadIdx.x & 31;
  const int wave = threadIdx.x >> 5;
  const int wm   = wave >> 2;              // 0..1
  const int wn   = wave & 3;               // 0..3
  const int m0   = blockIdx.x * 64  + wm * 32;
  const int n0   = blockIdx.y * 128 + wn * 32;
  const int lm   = lane & 15;
  const int hi   = lane >> 4;

  f32x8 acc[2][2] = {};

  int rowA0 = m0 + lm;       if (rowA0 >= M) rowA0 = M - 1;
  int rowA1 = m0 + 16 + lm;  if (rowA1 >= M) rowA1 = M - 1;
  int colB0 = n0 + lm;       if (colB0 >= N) colB0 = N - 1;
  int colB1 = n0 + 16 + lm;  if (colB1 >= N) colB1 = N - 1;

  const unsigned short* a0p = A + (size_t)rowA0 * K;
  const unsigned short* a1p = A + (size_t)rowA1 * K;
  const unsigned short* b0p = W + (size_t)colB0 * K;
  const unsigned short* b1p = W + (size_t)colB1 * K;

  for (int k0 = 0; k0 < K; k0 += 32) {
    AFrag af[2], bf[2];
#pragma unroll
    for (int v = 0; v < 8; ++v) {
      // A (16x32 bf16): lane holds row lane&15; K pairs per ISA layout
      int kA = k0 + ((v < 4) ? 0 : 16) + hi * 8 + (v & 3) * 2;
      af[0].u[v] = *(const unsigned int*)(a0p + kA);
      af[1].u[v] = *(const unsigned int*)(a1p + kA);
      // B (32x16 bf16): lane holds column lane&15; K pairs per ISA layout (same as A)
      int kB = k0 + ((v < 4) ? 0 : 16) + hi * 8 + (v & 3) * 2;
      bf[0].u[v] = *(const unsigned int*)(b0p + kB);
      bf[1].u[v] = *(const unsigned int*)(b1p + kB);
    }
#pragma unroll
    for (int i = 0; i < 2; ++i)
#pragma unroll
      for (int j = 0; j < 2; ++j)
        acc[i][j] = __builtin_amdgcn_wmma_f32_16x16x32_bf16(
            false, af[i].v, false, bf[j].v, (short)0, acc[i][j], false, false);
  }

  // Epilogue.  C/D layout: column = lane&15, row = (lane>>4)*8 + r
#pragma unroll
  for (int j = 0; j < 2; ++j) {
    int nc = n0 + j * 16 + lm;
    if (nc >= N) continue;
    float bv = bias ? bias[nc] : 0.f;
#pragma unroll
    for (int i = 0; i < 2; ++i) {
#pragma unroll
      for (int r = 0; r < 8; ++r) {
        int mr = m0 + i * 16 + hi * 8 + r;
        if (mr < M) {
          float val = acc[i][j][r] + bv;
          if (RELU) val = fmaxf(val, 0.f);
          if (WF32)  Of32[(size_t)mr * N + nc] = val;
          if (WBF16) Obf[(size_t)mr * N + nc]  = f2bf(val);
        }
      }
    }
  }
}

// ---------------------------------------------------------------------------
// fp32 -> bf16 conversion (weights)
// ---------------------------------------------------------------------------
__global__ void f32_to_bf16_kernel(const float* __restrict__ src,
                                   unsigned short* __restrict__ dst, int n)
{
  int i = blockIdx.x * blockDim.x + threadIdx.x;
  if (i < n) dst[i] = f2bf(src[i]);
}

__global__ void zero_out_kernel(float* p) { p[0] = 0.f; }

// ---------------------------------------------------------------------------
// Patch extraction + linear patch embedding + positional embedding
// ---------------------------------------------------------------------------
__global__ __launch_bounds__(256)
void embed_kernel(const float* __restrict__ x,
                  const float* __restrict__ W_lp, const float* __restrict__ b_lp,
                  const float* __restrict__ W_pos, const float* __restrict__ b_pos,
                  float* __restrict__ emb, float* __restrict__ gt)
{
  int n = blockIdx.x % NPATCH;
  int b = blockIdx.x / NPATCH;
  int gy = n / 10, gx = n % 10;
  __shared__ float patch[PDC];

  int p  = threadIdx.x;                    // 256 threads, one pixel each
  int py = p >> 4, px = p & 15;
  float val = x[((size_t)b * 160 + gy * 16 + py) * 160 + gx * 16 + px];
  patch[p] = val;
  gt[((size_t)b * NPATCH + n) * PDC + p] = val;
  __syncthreads();

  float xc = 0.05f + 0.1f * (float)gx;     // linspace(0.05, 0.95, 10)
  float yc = 0.05f + 0.1f * (float)gy;
  for (int d = threadIdx.x; d < DIMC; d += 256) {
    const float* wr = W_lp + (size_t)d * PDC;
    float s = b_lp[d];
    for (int q = 0; q < PDC; ++q) s += wr[q] * patch[q];
    float ce = W_pos[d * 2 + 0] * xc + W_pos[d * 2 + 1] * yc + b_pos[d];
    emb[((size_t)b * NPATCH + n) * DIMC + d] = s + ce;
  }
}

// ---------------------------------------------------------------------------
// Build X[i,s,b,:] = (s<=i ? E[s,b] : 0) + (s==i+1 ? query_token + ce[i] : 0)
// Row order m = (i*S + s)*B + b
// ---------------------------------------------------------------------------
__global__ __launch_bounds__(256)
void build_x_kernel(const float* __restrict__ emb,
                    const float* __restrict__ start_tok,
                    const float* __restrict__ query_tok,
                    const float* __restrict__ W_pos, const float* __restrict__ b_pos,
                    float* __restrict__ Xf, unsigned short* __restrict__ Xb)
{
  int idx = blockIdx.x;
  int b = idx % BATCH;  idx /= BATCH;
  int s = idx % SSEQ;
  int i = idx / SSEQ;
  size_t m = (size_t)blockIdx.x;
  int gy = i / 10, gx = i % 10;
  float xc = 0.05f + 0.1f * (float)gx;
  float yc = 0.05f + 0.1f * (float)gy;

  for (int d = threadIdx.x; d < DIMC; d += 256) {
    float v = 0.f;
    if (s <= i)
      v = (s == 0) ? start_tok[d]
                   : emb[((size_t)b * NPATCH + (s - 1)) * DIMC + d];
    if (s == i + 1)
      v += query_tok[d] + W_pos[d * 2] * xc + W_pos[d * 2 + 1] * yc + b_pos[d];
    Xf[m * DIMC + d] = v;
    Xb[m * DIMC + d] = f2bf(v);
  }
}

// ---------------------------------------------------------------------------
// Attention: one block per (i,b,h); valid keys s_key <= i+1.
// QKV is bf16 [m, 1536] (q|k|v at 0|512|1024, head h at h*64).
// K/V head rows staged into LDS via gfx1250 async global->LDS b128 copies
// (ASYNCcnt) when available; online softmax per query row.
// ---------------------------------------------------------------------------
__global__ __launch_bounds__(128)
void attention_kernel(const unsigned short* __restrict__ QKVb,
                      unsigned short* __restrict__ Obf)
{
  int blk = blockIdx.x;
  int h = blk % HEADSC; blk /= HEADSC;
  int b = blk % BATCH;
  int i = blk / BATCH;
  int nk = i + 2;                          // keys 0 .. i+1

  __shared__ unsigned short Ks[SSEQ * HDC];   // raw bf16
  __shared__ unsigned short Vs[SSEQ * HDC];

#if __has_builtin(__builtin_amdgcn_global_load_async_to_lds_b128) && \
    __has_builtin(__builtin_amdgcn_s_wait_asynccnt)
  // 16B per lane per async op, straight into LDS (no VGPR round trip).
  // One K/V head row = 64 bf16 = 128 B = 8 chunks of 16 B.
  for (int c = threadIdx.x; c < nk * 8; c += 128) {
    int s = c >> 3, cc = c & 7;
    size_t m = ((size_t)i * SSEQ + s) * BATCH + b;
    const unsigned short* kg = QKVb + m * 1536 + 512  + h * HDC + cc * 8;
    const unsigned short* vg = QKVb + m * 1536 + 1024 + h * HDC + cc * 8;
    __builtin_amdgcn_global_load_async_to_lds_b128(
        (__attribute__((address_space(1))) i32x4*)kg,
        (__attribute__((address_space(3))) i32x4*)((char*)Ks + c * 16), 0, 0);
    __builtin_amdgcn_global_load_async_to_lds_b128(
        (__attribute__((address_space(1))) i32x4*)vg,
        (__attribute__((address_space(3))) i32x4*)((char*)Vs + c * 16), 0, 0);
  }
  __builtin_amdgcn_s_wait_asynccnt(0);
  __syncthreads();
#else
  for (int idx = threadIdx.x; idx < nk * HDC; idx += 128) {
    int s = idx >> 6, d = idx & 63;
    size_t m = ((size_t)i * SSEQ + s) * BATCH + b;
    Ks[idx] = QKVb[m * 1536 + 512  + h * HDC + d];
    Vs[idx] = QKVb[m * 1536 + 1024 + h * HDC + d];
  }
  __syncthreads();
#endif

  const float scale = 0.125f;              // 1/sqrt(64)
  for (int s = threadIdx.x; s < SSEQ; s += 128) {
    size_t m = ((size_t)i * SSEQ + s) * BATCH + b;
    float q[HDC];
#pragma unroll
    for (int d = 0; d < HDC; ++d)
      q[d] = bf2f(QKVb[m * 1536 + h * HDC + d]) * scale;

    float mx = -3.0e38f, l = 0.f;
    float o[HDC];
#pragma unroll
    for (int d = 0; d < HDC; ++d) o[d] = 0.f;

    for (int k = 0; k < nk; ++k) {
      float sc = 0.f;
#pragma unroll
      for (int d = 0; d < HDC; ++d) sc += q[d] * bf2f(Ks[k * HDC + d]);
      float nm   = fmaxf(mx, sc);
      float corr = __expf(mx - nm);
      float p    = __expf(sc - nm);
      l = l * corr + p;
#pragma unroll
      for (int d = 0; d < HDC; ++d)
        o[d] = o[d] * corr + p * bf2f(Vs[k * HDC + d]);
      mx = nm;
    }
    float rl = 1.f / l;
#pragma unroll
    for (int d = 0; d < HDC; ++d)
      Obf[m * DIMC + h * HDC + d] = f2bf(o[d] * rl);
  }
}

// ---------------------------------------------------------------------------
// Fused residual add + LayerNorm.  One row per block (256 threads).
// ---------------------------------------------------------------------------
__global__ __launch_bounds__(256)
void residual_ln_kernel(const float* __restrict__ Xin, const float* __restrict__ R,
                        const float* __restrict__ g,   const float* __restrict__ bb,
                        float* __restrict__ Xout, unsigned short* __restrict__ Xbf)
{
  int row = blockIdx.x;
  __shared__ float red[256];
  __shared__ float vals[DIMC];
  const float* xr = Xin + (size_t)row * DIMC;
  const float* rr = R   + (size_t)row * DIMC;

  float s = 0.f;
  for (int d = threadIdx.x; d < DIMC; d += 256) {
    float v = xr[d] + rr[d];
    vals[d] = v;
    s += v;
  }
  red[threadIdx.x] = s; __syncthreads();
  for (int off = 128; off > 0; off >>= 1) {
    if (threadIdx.x < off) red[threadIdx.x] += red[threadIdx.x + off];
    __syncthreads();
  }
  float mean = red[0] * (1.f / DIMC);
  __syncthreads();

  float s2 = 0.f;
  for (int d = threadIdx.x; d < DIMC; d += 256) {
    float dv = vals[d] - mean;
    s2 += dv * dv;
  }
  red[threadIdx.x] = s2; __syncthreads();
  for (int off = 128; off > 0; off >>= 1) {
    if (threadIdx.x < off) red[threadIdx.x] += red[threadIdx.x + off];
    __syncthreads();
  }
  float rstd = rsqrtf(red[0] * (1.f / DIMC) + 1e-5f);

  for (int d = threadIdx.x; d < DIMC; d += 256) {
    float v = (vals[d] - mean) * rstd * g[d] + bb[d];
    Xout[(size_t)row * DIMC + d] = v;
    Xbf[(size_t)row * DIMC + d] = f2bf(v);
  }
}

// ---------------------------------------------------------------------------
// Gather y[n,b,:] = X[i=n, s=n+1, b, :]  (row order n*B+b), to bf16
// ---------------------------------------------------------------------------
__global__ __launch_bounds__(256)
void gather_y_kernel(const float* __restrict__ Xf, unsigned short* __restrict__ Yb)
{
  int nb = blockIdx.x;
  int b = nb % BATCH, n = nb / BATCH;
  size_t m = ((size_t)n * SSEQ + (n + 1)) * BATCH + b;
  for (int d = threadIdx.x; d < DIMC; d += 256)
    Yb[(size_t)nb * DIMC + d] = f2bf(Xf[m * DIMC + d]);
}

// ---------------------------------------------------------------------------
// VAE head / decoder: one block per (n,b), 64 threads.
// ---------------------------------------------------------------------------
__global__ __launch_bounds__(64)
void decoder_kernel(const float* __restrict__ cond, const float* __restrict__ noise,
                    const float* __restrict__ Wd_in, const float* __restrict__ bd_in,
                    const float* __restrict__ Wd,    const float* __restrict__ bd,
                    const float* __restrict__ Wd_out,const float* __restrict__ bd_out,
                    const float* __restrict__ gt,    float* __restrict__ out)
{
  int nb = blockIdx.x;
  int b = nb % BATCH, n = nb / BATCH;
  int t = threadIdx.x;
  __shared__ float samp[PDC];
  __shared__ float hbuf[64];
  __shared__ float red[64];

  float klp = 0.f;
#pragma unroll
  for (int j = 0; j < 4; ++j) {
    int p = j * 64 + t;
    float mu  = cond[(size_t)nb * 512 + p];
    float sig = cond[(size_t)nb * 512 + PDC + p];
    samp[p] = noise[(size_t)nb * PDC + p] * __expf(0.5f * sig) + mu;
    klp += 1.f + sig - __expf(sig);
  }
  __syncthreads();

  float hv = bd_in[t];
  for (int p = 0; p < PDC; ++p) hv += Wd_in[(size_t)t * PDC + p] * samp[p];
  hbuf[t] = hv;
  __syncthreads();

  for (int l = 0; l < 6; ++l) {
    float gv = bd[l * 64 + t];
    for (int k = 0; k < 64; ++k) gv += Wd[((size_t)l * 64 + t) * 64 + k] * hbuf[k];
    gv = fmaxf(gv, 0.f);
    __syncthreads();
    hbuf[t] += gv;
    __syncthreads();
  }

  float sq = 0.f;
#pragma unroll
  for (int j = 0; j < 4; ++j) {
    int p = j * 64 + t;
    float dv = bd_out[p];
    for (int k = 0; k < 64; ++k) dv += Wd_out[(size_t)p * 64 + k] * hbuf[k];
    float sp = 255.f / (1.f + __expf(-dv));
    float df = sp - gt[((size_t)b * NPATCH + n) * PDC + p];
    sq += df * df;
  }

  red[t] = sq; __syncthreads();
  for (int off = 32; off > 0; off >>= 1) {
    if (t < off) red[t] += red[t + off];
    __syncthreads();
  }
  float sq_total = red[0];
  __syncthreads();
  red[t] = klp; __syncthreads();
  for (int off = 32; off > 0; off >>= 1) {
    if (t < off) red[t] += red[t + off];
    __syncthreads();
  }
  if (t == 0) {
    float contrib = sq_total * (1.f / (BATCH * PDC))        // mse term
                  + 1e-4f * (-0.5f) * red[0];               // BETA * kl term
    atomicAdd(out, contrib);
  }
}

// ---------------------------------------------------------------------------
// Host orchestration.
// Input order = JAX pytree flatten (sorted keys) of {'noise','params','x'}:
//  0 noise | 1 W_lp | 2 W_out | 3 W_pos | 4 Wd | 5 Wd_in | 6 Wd_out
//  7 b_lp | 8 b_out | 9 b_pos | 10 bd | 11 bd_in | 12 bd_out
//  13 + L*12 : layers[L] {W1,W2,Wi,Wo,b1,b2,bi,bo,ln1_b,ln1_g,ln2_b,ln2_g}
//  61 query_token | 62 start_token | 63 x
// ---------------------------------------------------------------------------
extern "C" void kernel_launch(void* const* d_in, const int* in_sizes, int n_in,
                              void* d_out, int out_size, void* d_ws, size_t ws_size,
                              hipStream_t stream) {
  (void)in_sizes; (void)n_in; (void)out_size; (void)ws_size;

  const float* noise   = (const float*)d_in[0];
  const float* W_lp    = (const float*)d_in[1];
  const float* W_out   = (const float*)d_in[2];
  const float* W_pos   = (const float*)d_in[3];
  const float* Wd      = (const float*)d_in[4];
  const float* Wd_in   = (const float*)d_in[5];
  const float* Wd_out  = (const float*)d_in[6];
  const float* b_lp    = (const float*)d_in[7];
  const float* b_out   = (const float*)d_in[8];
  const float* b_pos   = (const float*)d_in[9];
  const float* bd      = (const float*)d_in[10];
  const float* bd_in   = (const float*)d_in[11];
  const float* bd_out  = (const float*)d_in[12];
  const float* query_t = (const float*)d_in[61];
  const float* start_t = (const float*)d_in[62];
  const float* x       = (const float*)d_in[63];

  const int LBASE = 13;
  auto LW1 = [&](int L){ return (const float*)d_in[LBASE + L*12 + 0]; };
  auto LW2 = [&](int L){ return (const float*)d_in[LBASE + L*12 + 1]; };
  auto LWi = [&](int L){ return (const float*)d_in[LBASE + L*12 + 2]; };
  auto LWo = [&](int L){ return (const float*)d_in[LBASE + L*12 + 3]; };
  auto Lb1 = [&](int L){ return (const float*)d_in[LBASE + L*12 + 4]; };
  auto Lb2 = [&](int L){ return (const float*)d_in[LBASE + L*12 + 5]; };
  auto Lbi = [&](int L){ return (const float*)d_in[LBASE + L*12 + 6]; };
  auto Lbo = [&](int L){ return (const float*)d_in[LBASE + L*12 + 7]; };
  auto Lln1b = [&](int L){ return (const float*)d_in[LBASE + L*12 + 8]; };
  auto Lln1g = [&](int L){ return (const float*)d_in[LBASE + L*12 + 9]; };
  auto Lln2b = [&](int L){ return (const float*)d_in[LBASE + L*12 + 10]; };
  auto Lln2g = [&](int L){ return (const float*)d_in[LBASE + L*12 + 11]; };

  // ---- workspace carve-up (256B aligned) ----
  char* base = (char*)d_ws;
  size_t cur = 0;
  auto alloc = [&](size_t bytes) {
    void* p = base + cur;
    cur += (bytes + 255) & ~(size_t)255;
    return p;
  };
  const size_t M = MROWS;
  float*          Xf    = (float*)         alloc(M * DIMC * 4);
  unsigned short* Xb    = (unsigned short*)alloc(M * DIMC * 2);
  unsigned short* QKVb  = (unsigned short*)alloc(M * 1536 * 2);  // bf16 q|k|v
  unsigned short* TMPB  = (unsigned short*)alloc(M * 1024 * 2);  // attn out / ffn1 out
  float*          TMPF  = (float*)         alloc(M * DIMC * 4);  // Wo / FFN2 out
  float*          EMB   = (float*)         alloc((size_t)BATCH * NPATCH * DIMC * 4);
  float*          GT    = (float*)         alloc((size_t)BATCH * NPATCH * PDC * 4);
  unsigned short* Yb    = (unsigned short*)alloc((size_t)NPATCH * BATCH * DIMC * 2);
  float*          COND  = (float*)         alloc((size_t)NPATCH * BATCH * 512 * 4);
  unsigned short* WiB[4], *WoB[4], *W1B[4], *W2B[4];
  for (int L = 0; L < 4; ++L) {
    WiB[L] = (unsigned short*)alloc((size_t)1536 * 512 * 2);
    WoB[L] = (unsigned short*)alloc((size_t)512  * 512 * 2);
    W1B[L] = (unsigned short*)alloc((size_t)1024 * 512 * 2);
    W2B[L] = (unsigned short*)alloc((size_t)512  * 1024 * 2);
  }
  unsigned short* WoutB = (unsigned short*)alloc((size_t)512 * 512 * 2);

  // ---- init + weight conversion ----
  zero_out_kernel<<<1, 1, 0, stream>>>((float*)d_out);
  auto cvt = [&](const float* src, unsigned short* dst, int n) {
    f32_to_bf16_kernel<<<(n + 255) / 256, 256, 0, stream>>>(src, dst, n);
  };
  for (int L = 0; L < 4; ++L) {
    cvt(LWi(L), WiB[L], 1536 * 512);
    cvt(LWo(L), WoB[L], 512  * 512);
    cvt(LW1(L), W1B[L], 1024 * 512);
    cvt(LW2(L), W2B[L], 512  * 1024);
  }
  cvt(W_out, WoutB, 512 * 512);

  // ---- embedding + X construction ----
  embed_kernel<<<BATCH * NPATCH, 256, 0, stream>>>(x, W_lp, b_lp, W_pos, b_pos,
                                                   EMB, GT);
  build_x_kernel<<<(int)M, 256, 0, stream>>>(EMB, start_t, query_t, W_pos, b_pos,
                                             Xf, Xb);

  // ---- transformer layers ----
  const int gm = (int)((M + 63) / 64);
  for (int L = 0; L < 4; ++L) {
    // QKV: [M,512] x [1536,512]^T -> bf16 [M,1536]  (halves QKV traffic)
    wmma_gemm_bf16<0, 0, 1><<<dim3(gm, 1536 / 128), 256, 0, stream>>>(
        Xb, WiB[L], Lbi(L), nullptr, QKVb, (int)M, 1536, 512);
    // attention -> bf16 o[M,512]
    attention_kernel<<<NPATCH * BATCH * HEADSC, 128, 0, stream>>>(QKVb, TMPB);
    // Wo: [M,512] x [512,512]^T -> f32
    wmma_gemm_bf16<0, 1, 0><<<dim3(gm, 512 / 128), 256, 0, stream>>>(
        TMPB, WoB[L], Lbo(L), TMPF, nullptr, (int)M, 512, 512);
    // X = LN(X + o)
    residual_ln_kernel<<<(int)M, 256, 0, stream>>>(TMPF, Xf, Lln1g(L), Lln1b(L),
                                                   Xf, Xb);
    // FFN1: relu([M,512] x [1024,512]^T) -> bf16 [M,1024]
    wmma_gemm_bf16<1, 0, 1><<<dim3(gm, 1024 / 128), 256, 0, stream>>>(
        Xb, W1B[L], Lb1(L), nullptr, TMPB, (int)M, 1024, 512);
    // FFN2: [M,1024] x [512,1024]^T -> f32 [M,512]
    wmma_gemm_bf16<0, 1, 0><<<dim3(gm, 512 / 128), 256, 0, stream>>>(
        TMPB, W2B[L], Lb2(L), TMPF, nullptr, (int)M, 512, 1024);
    // X = LN(X + f)
    residual_ln_kernel<<<(int)M, 256, 0, stream>>>(TMPF, Xf, Lln2g(L), Lln2b(L),
                                                   Xf, Xb);
  }

  // ---- head: cond = y @ W_out^T + b_out ----
  gather_y_kernel<<<NPATCH * BATCH, 256, 0, stream>>>(Xf, Yb);
  wmma_gemm_bf16<0, 1, 0><<<dim3((NPATCH * BATCH + 63) / 64, 512 / 128),
                            256, 0, stream>>>(
      Yb, WoutB, b_out, COND, nullptr, NPATCH * BATCH, 512, 512);

  // ---- decoder + loss reduction ----
  decoder_kernel<<<NPATCH * BATCH, 64, 0, stream>>>(
      COND, noise, Wd_in, bd_in, Wd, bd, Wd_out, bd_out, GT, (float*)d_out);
}
